// LexiconAdapter_58248346468936
// MI455X (gfx1250) — compile-verified, matrix operand order checked by reading, not run on hardware
//
#include <hip/hip_runtime.h>
#include <hip/hip_bf16.h>
#include <math.h>

// ---------------------------------------------------------------------------
// CDNA5 WMMA types (wave32): D(16x16 f32) = A(16x32 bf16) x B(32x16 bf16) + C
// ---------------------------------------------------------------------------
typedef __attribute__((ext_vector_type(16))) __bf16 v16bf;
typedef __attribute__((ext_vector_type(8)))  float  v8f;

__device__ __forceinline__ unsigned short f2bf(float f) {
    union { float f; unsigned u; } x; x.f = f;
    unsigned r = x.u + 0x7FFFu + ((x.u >> 16) & 1u);   // round-to-nearest-even
    return (unsigned short)(r >> 16);
}

// CDNA5 async memory->LDS copy (16B per lane), tracked by ASYNCcnt.
__device__ __forceinline__ void async_b128(unsigned lds_byte_addr,
                                           const unsigned short* g) {
    asm volatile("global_load_async_to_lds_b128 %0, %1, off"
                 :: "v"(lds_byte_addr),
                    "v"(reinterpret_cast<unsigned long long>(g))
                 : "memory");
}

// ---------------------------------------------------------------------------
// Pre-pass kernels (bandwidth-trivial, run once)
// ---------------------------------------------------------------------------
// fp32 [M,K] -> bf16 [M,Kp], zero-padded in K
__global__ void cvt_pad_kernel(const float* __restrict__ in,
                               unsigned short* __restrict__ out,
                               int K, int Kp, long long total) {
    long long i = (long long)blockIdx.x * 256 + threadIdx.x;
    if (i >= total) return;
    long long r = i / Kp;
    int k = (int)(i - r * Kp);
    out[i] = (k < K) ? f2bf(in[r * K + k]) : (unsigned short)0;
}

// fp32 [K,N] -> bf16 transposed [N,Kp], zero-padded in K
__global__ void transpose_cvt_kernel(const float* __restrict__ in,
                                     unsigned short* __restrict__ out,
                                     int K, int N, int Kp) {
    long long i = (long long)blockIdx.x * 256 + threadIdx.x;
    if (i >= (long long)N * Kp) return;
    int n = (int)(i / Kp);
    int k = (int)(i - (long long)n * Kp);
    out[i] = (k < K) ? f2bf(in[(size_t)k * N + n]) : (unsigned short)0;
}

// ---------------------------------------------------------------------------
// Double-buffered async WMMA GEMM: Out[M,N] = act(A[M,Kp] * Bt[N,Kp]^T + bias)
//   A bf16 row-major [M,Kp], Bt bf16 pre-transposed [N,Kp]; Kp % 32 == 0,
//   M % 128 == 0, N % 128 == 0 (all true here).
//   Block tile 128(M) x 128(N); 8 waves (wave32), wave grid 4(M) x 2(N),
//   each wave a 2(M) x 4(N) grid of 16x16 fragments -> 8 v_wmma per K-step
//   against 12 ds_load_b128 (A frags reused over 4 N-frags).
//   Staging: branchless double buffer, 4 global_load_async_to_lds_b128 per
//   thread per K-step, drained with s_wait_asynccnt.
// ---------------------------------------------------------------------------
template<int DO_TANH, int OUT_IS_BF16, int HAS_BIAS>
__global__ __launch_bounds__(256)
void gemm_wmma_async(const unsigned short* __restrict__ A,
                     const unsigned short* __restrict__ Bt,
                     const float* __restrict__ bias,
                     void* __restrict__ Out,
                     int M, int N, int Kp) {
    // LDS layout (ushort elems): A bufs [2][128][40], then B bufs [2][128][40]
    constexpr int AR = 40;                  // padded K-row (80 B, 16B-aligned cols)
    constexpr int ABUF = 128 * AR;          // 5120 elems
    constexpr int BBASE = 2 * ABUF;         // 10240 elems
    constexpr int BBUF = 128 * AR;          // 5120 elems
    __shared__ __align__(16) unsigned short lds[2 * ABUF + 2 * BBUF];   // 40 KB

    const unsigned ldsb = (unsigned)reinterpret_cast<unsigned long long>(&lds[0]);

    const int m0   = blockIdx.y * 128;
    const int n0   = blockIdx.x * 128;
    const int tid  = threadIdx.x;
    const int lane = tid & 31;
    const int wave = tid >> 5;
    const int waveM = wave & 3;      // 4 x 32 rows
    const int waveN = wave >> 2;     // 2 x 64 cols
    const int half  = lane >> 4;     // 0: K {kb..kb+7, kb+16..kb+23} kb=0; 1: kb=8
    const int l16   = lane & 15;
    const int kb    = half * 8;

    // per-thread staging coordinates: 128x32 tile = 256 threads x 2 x 16B
    const int srow = tid >> 2, skc = (tid & 3) * 8;        // rows 0..63
    const unsigned a_lds0 = ldsb + (unsigned)(srow * AR + skc) * 2;
    const unsigned a_lds1 = ldsb + (unsigned)((srow + 64) * AR + skc) * 2;
    const unsigned b_lds0 = ldsb + (unsigned)(BBASE + srow * AR + skc) * 2;
    const unsigned b_lds1 = ldsb + (unsigned)(BBASE + (srow + 64) * AR + skc) * 2;
    const unsigned short* a_g0 = A  + (size_t)(m0 + srow) * Kp + skc;
    const unsigned short* a_g1 = A  + (size_t)(m0 + srow + 64) * Kp + skc;
    const unsigned short* b_g0 = Bt + (size_t)(n0 + srow) * Kp + skc;
    const unsigned short* b_g1 = Bt + (size_t)(n0 + srow + 64) * Kp + skc;

    auto issue_tile = [&](int kk, int buf) {
        const unsigned ao = (unsigned)(buf * ABUF) * 2;
        const unsigned bo = (unsigned)(buf * BBUF) * 2;
        async_b128(a_lds0 + ao, a_g0 + kk);
        async_b128(a_lds1 + ao, a_g1 + kk);
        async_b128(b_lds0 + bo, b_g0 + kk);
        async_b128(b_lds1 + bo, b_g1 + kk);
    };

    v8f acc[2][4];
    #pragma unroll
    for (int i = 0; i < 2; ++i)
        #pragma unroll
        for (int j = 0; j < 4; ++j)
            #pragma unroll
            for (int e = 0; e < 8; ++e) acc[i][j][e] = 0.0f;

    const int nk = Kp >> 5;
    const int kklast = (nk - 1) << 5;
    issue_tile(0, 0);

    for (int ik = 0; ik < nk; ++ik) {
        const int buf = ik & 1;
        // branchless prefetch of next tile (clamped; final extra load is dead)
        int kknext = (ik + 1) << 5;
        if (kknext > kklast) kknext = kklast;
        issue_tile(kknext, buf ^ 1);
        // <=4 outstanding -> the 4 ops of tile ik have landed (in-order dones)
        asm volatile("s_wait_asynccnt 0x4" ::: "memory");
        __syncthreads();

        union FragU { v16bf v; uint4 u[2]; };
        FragU a[2], b[4];
        const int abase = buf * ABUF;
        const int bbase = BBASE + buf * BBUF;
        #pragma unroll
        for (int fm = 0; fm < 2; ++fm) {
            int row = waveM * 32 + fm * 16 + l16;
            a[fm].u[0] = *(const uint4*)&lds[abase + row * AR + kb];
            a[fm].u[1] = *(const uint4*)&lds[abase + row * AR + kb + 16];
        }
        #pragma unroll
        for (int fn = 0; fn < 4; ++fn) {
            int col = waveN * 64 + fn * 16 + l16;
            b[fn].u[0] = *(const uint4*)&lds[bbase + col * AR + kb];
            b[fn].u[1] = *(const uint4*)&lds[bbase + col * AR + kb + 16];
        }
        #pragma unroll
        for (int fm = 0; fm < 2; ++fm)
            #pragma unroll
            for (int fn = 0; fn < 4; ++fn)
                acc[fm][fn] = __builtin_amdgcn_wmma_f32_16x16x32_bf16(
                    false, a[fm].v, false, b[fn].v,
                    (short)0, acc[fm][fn], false, false);

        __syncthreads();   // everyone done reading buf before it is refilled
    }

    // epilogue: lanes 0..15 -> N=l16, M=r; lanes 16..31 -> M=8+r
    #pragma unroll
    for (int fm = 0; fm < 2; ++fm) {
        #pragma unroll
        for (int fn = 0; fn < 4; ++fn) {
            int col = n0 + waveN * 64 + fn * 16 + l16;
            float bv = HAS_BIAS ? bias[col] : 0.0f;
            #pragma unroll
            for (int r = 0; r < 8; ++r) {
                int rowm = m0 + waveM * 32 + fm * 16 + half * 8 + r;
                float v = acc[fm][fn][r] + bv;
                if (DO_TANH) v = tanhf(v);
                if (OUT_IS_BF16)
                    ((unsigned short*)Out)[(size_t)rowm * N + col] = f2bf(v);
                else
                    ((float*)Out)[(size_t)rowm * N + col] = v;
            }
        }
    }
}

// ---------------------------------------------------------------------------
// Fused per-token: bilinear attention + masked softmax + aggregation + LN
// One block (256 threads) per token; H = 768 = 256 * 3.
// ---------------------------------------------------------------------------
__device__ __forceinline__ float block_sum(float v, float* sm) {
    int t = threadIdx.x;
    sm[t] = v; __syncthreads();
    #pragma unroll
    for (int s = 128; s > 0; s >>= 1) {
        if (t < s) sm[t] += sm[t + s];
        __syncthreads();
    }
    float r = sm[0];
    __syncthreads();
    return r;
}

__global__ __launch_bounds__(256)
void attn_ln_kernel(const float* __restrict__ x,      // [BS,768] layer_output
                    const float* __restrict__ we,     // [BS,5,768]
                    const float* __restrict__ q,      // [BS,768]
                    const int*   __restrict__ mask,   // [BS,5]
                    const float* __restrict__ gamma,
                    const float* __restrict__ beta,
                    float* __restrict__ out) {
    const int token = blockIdx.x;
    const int t = threadIdx.x;
    __shared__ float sm[256];
    __shared__ float sval[5];

    const float* xr = x  + (size_t)token * 768;
    const float* qr = q  + (size_t)token * 768;
    const float* wr = we + (size_t)token * 5 * 768;
    const int*   mr = mask + (size_t)token * 5;

    float xv[3], qv[3], wv[5][3];
    #pragma unroll
    for (int i = 0; i < 3; ++i) {
        int h = t + i * 256;
        xv[i] = xr[h];
        qv[i] = qr[h];
        #pragma unroll
        for (int w = 0; w < 5; ++w) wv[w][i] = wr[w * 768 + h];
    }

    #pragma unroll
    for (int w = 0; w < 5; ++w) {
        float p = qv[0] * wv[w][0] + qv[1] * wv[w][1] + qv[2] * wv[w][2];
        float s = block_sum(p, sm);
        if (t == 0) sval[w] = s;
    }
    __syncthreads();

    float a[5], mx = -3.4e38f;
    #pragma unroll
    for (int w = 0; w < 5; ++w) {
        float v = sval[w] + (mr[w] ? 0.0f : -10000.0f);
        a[w] = v;
        mx = fmaxf(mx, v);
    }
    float den = 0.0f;
    #pragma unroll
    for (int w = 0; w < 5; ++w) { a[w] = __expf(a[w] - mx); den += a[w]; }
    float inv = 1.0f / den;

    float hv[3];
    #pragma unroll
    for (int i = 0; i < 3; ++i) {
        float agg = 0.0f;
        #pragma unroll
        for (int w = 0; w < 5; ++w) agg += a[w] * wv[w][i];
        hv[i] = xv[i] + agg * inv;
    }

    float mu = block_sum(hv[0] + hv[1] + hv[2], sm) * (1.0f / 768.0f);
    float d0 = hv[0] - mu, d1 = hv[1] - mu, d2 = hv[2] - mu;
    float var = block_sum(d0 * d0 + d1 * d1 + d2 * d2, sm) * (1.0f / 768.0f);
    float rstd = rsqrtf(var + 1e-12f);

    float dv[3] = {d0, d1, d2};
    #pragma unroll
    for (int i = 0; i < 3; ++i) {
        int h = t + i * 256;
        out[(size_t)token * 768 + h] = dv[i] * rstd * gamma[h] + beta[h];
    }
}

// ---------------------------------------------------------------------------
// Launch
// ---------------------------------------------------------------------------
extern "C" void kernel_launch(void* const* d_in, const int* in_sizes, int n_in,
                              void* d_out, int out_size, void* d_ws, size_t ws_size,
                              hipStream_t stream) {
    const float* layer_output = (const float*)d_in[0];  // [16,512,768]
    const float* word_emb     = (const float*)d_in[1];  // [16,512,5,200]
    const int*   word_mask    = (const int*)d_in[2];    // [16,512,5] (jax int32)
    const float* W1           = (const float*)d_in[3];  // [200,768]
    const float* b1           = (const float*)d_in[4];
    const float* W2           = (const float*)d_in[5];  // [768,768]
    const float* b2           = (const float*)d_in[6];
    const float* attn_W       = (const float*)d_in[7];  // [768,768]
    const float* ln_gamma     = (const float*)d_in[8];
    const float* ln_beta      = (const float*)d_in[9];
    float* out = (float*)d_out;

    constexpr int D = 200, Dp = 224, H = 768;
    constexpr int BS  = 16 * 512;       // 8192 tokens
    constexpr int BSW = BS * 5;         // 40960 word rows

    char* wsb = (char*)d_ws;
    size_t off = 0;
    auto carve = [&](size_t bytes) -> void* {
        void* p = wsb + off;
        off += (bytes + 255) & ~(size_t)255;
        return p;
    };
    unsigned short* W1t = (unsigned short*)carve((size_t)H * Dp * 2);   // [768,224]
    unsigned short* W2t = (unsigned short*)carve((size_t)H * H * 2);    // [768,768]
    unsigned short* AWt = (unsigned short*)carve((size_t)H * H * 2);    // [768,768]
    unsigned short* Aw  = (unsigned short*)carve((size_t)BSW * Dp * 2); // padded emb
    unsigned short* Xb  = (unsigned short*)carve((size_t)BS * H * 2);   // bf16 x
    unsigned short* tbf = (unsigned short*)carve((size_t)BSW * H * 2);  // tanh out
    float*          we  = (float*)carve((size_t)BSW * H * 4);
    float*          q   = (float*)carve((size_t)BS  * H * 4);

    // --- pre-passes: convert + pad A panels, transpose + convert weights ---
    {
        long long n = (long long)BSW * Dp;
        cvt_pad_kernel<<<(unsigned)((n + 255) / 256), 256, 0, stream>>>(
            word_emb, Aw, D, Dp, n);
    }
    {
        long long n = (long long)BS * H;
        cvt_pad_kernel<<<(unsigned)((n + 255) / 256), 256, 0, stream>>>(
            layer_output, Xb, H, H, n);
    }
    transpose_cvt_kernel<<<(H * Dp + 255) / 256, 256, 0, stream>>>(W1, W1t, D, H, Dp);
    transpose_cvt_kernel<<<(H * H + 255) / 256, 256, 0, stream>>>(W2, W2t, H, H, H);
    transpose_cvt_kernel<<<(H * H + 255) / 256, 256, 0, stream>>>(attn_W, AWt, H, H, H);

    dim3 blk(256);
    // GEMM1 + bias + tanh -> tbf (bf16): [40960,224] x [224,768]
    gemm_wmma_async<1, 1, 1><<<dim3(H / 128, BSW / 128), blk, 0, stream>>>(
        Aw, W1t, b1, tbf, BSW, H, Dp);
    // GEMM2 + bias -> we (f32): [40960,768] x [768,768]
    gemm_wmma_async<0, 0, 1><<<dim3(H / 128, BSW / 128), blk, 0, stream>>>(
        tbf, W2t, b2, we, BSW, H, H);
    // GEMM3 -> q (f32): [8192,768] x [768,768]
    gemm_wmma_async<0, 0, 0><<<dim3(H / 128, BS / 128), blk, 0, stream>>>(
        Xb, AWt, nullptr, q, BS, H, H);
    // fused attention + softmax + aggregation + residual + LayerNorm
    attn_ln_kernel<<<BS, 256, 0, stream>>>(layer_output, we, q, word_mask,
                                           ln_gamma, ln_beta, out);
}
